// NT_Xent_20478404067977
// MI455X (gfx1250) — matrix operand order repaired and checked by reading, not compiled, not run on hardware
//
#include <hip/hip_runtime.h>
#include <hip/hip_bf16.h>

typedef _Float16 v8h  __attribute__((ext_vector_type(8)));
typedef _Float16 v16h __attribute__((ext_vector_type(16)));
typedef float    v8f  __attribute__((ext_vector_type(8)));

#define NROW   4096
#define DIM    512
#define TWO_N  8192
#define INV_T  2.0f           /* 1 / TEMP */
#define MARG_T 0.2f           /* MARGIN / TEMP */
#define MB     2.5f           /* fixed softmax max-bound: logits <= 2.0 by cosine<=1 */
#define NCHUNK 8              /* column split */
#define LDA    520            /* padded LDS row stride (halves) */

/* CDNA5 async memory->LDS copy (ASYNCcnt-tracked), GVS addressing:
   mem = SADDR(64b SGPR) + VADDR(32b unsigned byte offset); dsaddr = lds addr VGPR. */
__device__ __forceinline__ void async_ld_b128(uint32_t lds, uint32_t voff,
                                              const void* sbase) {
  asm volatile("global_load_async_to_lds_b128 %0, %1, %2"
               :: "v"(lds), "v"(voff), "s"(sbase)
               : "memory");
}
__device__ __forceinline__ void wait_async0() {
  asm volatile("s_wait_asynccnt 0x0" ::: "memory");
}
__device__ __forceinline__ uint32_t lds_addr32(const void* p) {
  return (uint32_t)(uintptr_t)p;   /* generic LDS addr: low 32 bits == DS offset */
}

/* ---------------- Kernel 1: fp32 row-normalize -> f16, one wave per row -------- */
__global__ void normalize_rows(const float* __restrict__ zi,
                               const float* __restrict__ zj,
                               _Float16* __restrict__ zh) {
  const int row  = (blockIdx.x * blockDim.x + threadIdx.x) >> 5;
  const int lane = threadIdx.x & 31;
  const float* src = (row < NROW) ? (zi + (size_t)row * DIM)
                                  : (zj + (size_t)(row - NROW) * DIM);
  float v[16];
  float ss = 0.f;
#pragma unroll
  for (int i = 0; i < 16; ++i) { v[i] = src[i * 32 + lane]; ss += v[i] * v[i]; }
#pragma unroll
  for (int off = 16; off > 0; off >>= 1) ss += __shfl_xor(ss, off);
  const float inv = rsqrtf(fmaxf(ss, 1e-24f));
#pragma unroll
  for (int i = 0; i < 16; ++i)
    zh[(size_t)row * DIM + i * 32 + lane] = (_Float16)(v[i] * inv);
}

/* ---------------- Kernel 2: WMMA Gram tiles + fused bounded softmax ------------ */
/* grid = 512: blockIdx>>3 = row block (128 rows), blockIdx&7 = 1024-col chunk.
   Each wave: 16 rows x 32 cols per step (2 WMMA tiles sharing one A fragment).
   A (130 KB) LDS-resident; B double-buffered and filled by async-to-LDS DMA.
   Fragment registers are double-buffered so ds_load latency overlaps WMMA.      */
__global__ __launch_bounds__(256, 1)
void gram_softmax_partial(const _Float16* __restrict__ zh,
                          float* __restrict__ ps,   /* [8192][8] sum exp(lg-MB) */
                          float* __restrict__ pp) { /* [8192][8] positive logit */
  __shared__ _Float16 As[128 * LDA];
  __shared__ _Float16 Bs[2 * 32 * LDA];

  const int tid     = threadIdx.x;
  const int lane    = tid & 31;
  const int wave    = tid >> 5;
  const int rowBase = (blockIdx.x >> 3) * 128;
  const int chunk   = blockIdx.x & (NCHUNK - 1);
  const int colBase = chunk * (TWO_N / NCHUNK);

  /* stage A: 128 rows x 512 halves -> LDS via async DMA (32 x b128 per thread) */
  for (int i = tid; i < 128 * 64; i += 256) {
    const int r = i >> 6, c = (i & 63) << 3;
    async_ld_b128(lds_addr32(&As[r * LDA + c]),
                  (uint32_t)(((rowBase + r) * DIM + c) * 2), zh);
  }

  auto stageB = [&](int jj, int buf) {       /* 32 cols x 1KB = 32 KB / step */
#pragma unroll
    for (int t = 0; t < 8; ++t) {
      const int uid = tid + t * 256;
      const int r = uid >> 6, c = (uid & 63) << 3;
      async_ld_b128(lds_addr32(&Bs[buf * 32 * LDA + r * LDA + c]),
                    (uint32_t)(((colBase + jj * 32 + r) * DIM + c) * 2), zh);
    }
  };

  stageB(0, 0);
  wait_async0();
  __syncthreads();

  float ss_[8], po[8];
#pragma unroll
  for (int v = 0; v < 8; ++v) { ss_[v] = 0.f; po[v] = 0.f; }

  const int ha      = lane >> 4;
  const int col16   = lane & 15;
  const int arow    = wave * 16 + col16;
  const int rowTile = rowBase + wave * 16;

  /* wave-uniform special-tile indices (16-col tile units within this chunk) */
  const int posBase = (rowTile + NROW) & (TWO_N - 1);
  const int pjPos  = (posBase >= colBase && posBase < colBase + TWO_N / NCHUNK)
                       ? ((posBase - colBase) >> 4) : -1;
  const int pjDiag = (rowTile >= colBase && rowTile < colBase + TWO_N / NCHUNK)
                       ? ((rowTile - colBase) >> 4) : -1;

  auto consume = [&](const v8f& cc, int tIdx) {
    if (tIdx == pjPos || tIdx == pjDiag) {   /* rare fixup tiles (<=2 of 64) */
      const bool isPos = (tIdx == pjPos);
#pragma unroll
      for (int v = 0; v < 8; ++v) {
        float lg = cc[v] * INV_T;
        const bool hit = (col16 == (v + (ha << 3)));  /* tile diagonal */
        if (isPos && hit) { lg -= MARG_T; po[v] = lg; }
        float e = __expf(lg - MB);
        if (!isPos && hit) e = 0.f;          /* self-similarity masked out */
        ss_[v] += e;
      }
    } else {                                 /* hot path: 1 exp + 1 add / elem */
#pragma unroll
      for (int v = 0; v < 8; ++v) ss_[v] += __expf(cc[v] * INV_T - MB);
    }
  };

  for (int jj = 0; jj < 32; ++jj) {
    if (jj + 1 < 32) stageB(jj + 1, (jj + 1) & 1);   /* async fill next buffer */
    const _Float16* Bp = &Bs[(jj & 1) * 32 * LDA];

    /* fragment load for k-chunk kt into a given register set */
    auto loadFrag = [&](int kt, v16h& a, v16h& b0, v16h& b1) {
      const int k0 = kt * 32;
      const _Float16* ab = &As[arow * LDA + k0 + ha * 16];
      v8h alo = *(const v8h*)(ab);
      v8h ahi = *(const v8h*)(ab + 8);
      a = __builtin_shufflevector(alo, ahi, 0,1,2,3,4,5,6,7,8,9,10,11,12,13,14,15);

      const _Float16* bb0 = &Bp[col16 * LDA + k0 + ha * 16];
      v8h b0l = *(const v8h*)(bb0);
      v8h b0h = *(const v8h*)(bb0 + 8);
      b0 = __builtin_shufflevector(b0l, b0h, 0,1,2,3,4,5,6,7,8,9,10,11,12,13,14,15);

      const _Float16* bb1 = &Bp[(col16 + 16) * LDA + k0 + ha * 16];
      v8h b1l = *(const v8h*)(bb1);
      v8h b1h = *(const v8h*)(bb1 + 8);
      b1 = __builtin_shufflevector(b1l, b1h, 0,1,2,3,4,5,6,7,8,9,10,11,12,13,14,15);
    };

    v8f c0 = {0.f, 0.f, 0.f, 0.f, 0.f, 0.f, 0.f, 0.f};
    v8f c1 = {0.f, 0.f, 0.f, 0.f, 0.f, 0.f, 0.f, 0.f};

    /* software-pipelined k-loop: distinct register sets per parity so the
       scheduler can keep next-step ds_loads in flight during WMMA (partial
       s_wait_dscnt instead of full drains). */
    v16h aF[2], b0F[2], b1F[2];
    loadFrag(0, aF[0], b0F[0], b1F[0]);
#pragma unroll
    for (int kt = 0; kt < 16; ++kt) {
      const int cur = kt & 1;
      const int nxt = cur ^ 1;
      if (kt + 1 < 16) loadFrag(kt + 1, aF[nxt], b0F[nxt], b1F[nxt]);
      c0 = __builtin_amdgcn_wmma_f32_16x16x32_f16(false, aF[cur], false, b0F[cur],
                                                  (short)0, c0, false, false);
      c1 = __builtin_amdgcn_wmma_f32_16x16x32_f16(false, aF[cur], false, b1F[cur],
                                                  (short)0, c1, false, false);
    }

    consume(c0, jj * 2);
    consume(c1, jj * 2 + 1);

    wait_async0();           /* this wave's DMA into next buffer has landed */
    __syncthreads();         /* all waves' DMA visible before next iteration */
  }

  /* plain sums across the 16 column-residue lanes (shared fixed max-bound) */
#pragma unroll
  for (int off = 1; off < 16; off <<= 1) {
#pragma unroll
    for (int v = 0; v < 8; ++v) {
      ss_[v] += __shfl_xor(ss_[v], off);
      po[v]  += __shfl_xor(po[v], off);
    }
  }
  if ((lane & 15) == 0) {
#pragma unroll
    for (int v = 0; v < 8; ++v) {
      const int row = rowTile + (ha << 3) + v;
      const int idx = row * NCHUNK + chunk;
      ps[idx] = ss_[v];
      pp[idx] = po[v];
    }
  }
}

/* ---------------- Kernel 3: deterministic merge + mean ------------------------- */
__global__ void finalize_loss(const float* __restrict__ ps,
                              const float* __restrict__ pp,
                              float* __restrict__ out) {
  __shared__ float red[256];
  const int t = threadIdx.x;
  float acc = 0.f;
  for (int r = t; r < TWO_N; r += 256) {
    float S = 0.f, P = 0.f;
#pragma unroll
    for (int c = 0; c < NCHUNK; ++c) {
      S += ps[r * NCHUNK + c];
      P += pp[r * NCHUNK + c];
    }
    acc += (MB + __logf(S)) - P;            /* logsumexp - positive logit */
  }
  red[t] = acc;
  __syncthreads();
  for (int s = 128; s > 0; s >>= 1) {
    if (t < s) red[t] += red[t + s];
    __syncthreads();
  }
  if (t == 0) out[0] = red[0] / (float)TWO_N;
}

extern "C" void kernel_launch(void* const* d_in, const int* in_sizes, int n_in,
                              void* d_out, int out_size, void* d_ws, size_t ws_size,
                              hipStream_t stream) {
  const float* zi = (const float*)d_in[0];
  const float* zj = (const float*)d_in[1];

  _Float16* zh = (_Float16*)d_ws;                       /* 8192x512 f16 = 8 MB */
  float* ps = (float*)((char*)d_ws + (size_t)TWO_N * DIM * 2);
  float* pp = ps + TWO_N * NCHUNK;

  normalize_rows<<<TWO_N / 8, 256, 0, stream>>>(zi, zj, zh);
  gram_softmax_partial<<<(TWO_N / 128) * NCHUNK, 256, 0, stream>>>(zh, ps, pp);
  finalize_loss<<<1, 256, 0, stream>>>(ps, pp, (float*)d_out);
}
